// Custom_GroupQueryAttention_48395691491536
// MI455X (gfx1250) — compile-verified, matrix operand order checked by reading, not run on hardware
//
#include <hip/hip_runtime.h>

// ---- problem constants (match reference) ----
#define TT 2048
#define DD 4096
#define HH 32
#define GG 8
#define HDD 128
#define NQKV 6144   // H*HD + 2*G*HD

typedef __attribute__((ext_vector_type(16))) __bf16 v16bf;
typedef __attribute__((ext_vector_type(8)))  __bf16 v8bf;
typedef __attribute__((ext_vector_type(8)))  float  v8f;
typedef __attribute__((ext_vector_type(4)))  int    v4i;
typedef unsigned short u16;

#if __has_builtin(__builtin_amdgcn_global_load_async_to_lds_b128) && \
    __has_builtin(__builtin_amdgcn_s_wait_asynccnt)
#define HAVE_ASYNC 1
#else
#define HAVE_ASYNC 0
#endif

// ---- helpers ----
__device__ __forceinline__ u16 f32_to_bf16(float f) {
  union { float f; unsigned int u; } c; c.f = f;
  unsigned int u = c.u;
  unsigned int r = (u + 0x7FFFu + ((u >> 16) & 1u)) >> 16;   // RNE
  return (u16)r;
}

__device__ __forceinline__ v8f vzero8() {
  v8f z;
#pragma unroll
  for (int i = 0; i < 8; ++i) z[i] = 0.0f;
  return z;
}

// Load a 16-bit A/B WMMA fragment (16x32 / 32x16) for one lane.
// ISA layout (CDNA5 ISA, 16-bit A 16x32): lane-half selects K-split:
//   lanes 0-15 : K = {0..7} and {16..23};  lanes 16-31 : K = {8..15} and {24..31}
// 'rowptr' points at this lane's row/column K-base (contiguous K).
__device__ __forceinline__ v16bf ld_ab_frag(const u16* rowptr, int lh) {
  union { v16bf v; v8bf h[2]; } u;
  u.h[0] = *(const v8bf*)(rowptr + lh * 8);
  u.h[1] = *(const v8bf*)(rowptr + 16 + lh * 8);
  return u.v;
}

__device__ __forceinline__ v8f wmma_bf16(v16bf a, v16bf b, v8f c) {
  // D = A x B + C, 16x16x32 bf16 -> f32, wave32
  return __builtin_amdgcn_wmma_f32_16x16x32_bf16(false, a, false, b,
                                                 (short)0, c, false, false);
}

// 16-byte global -> LDS copy; async (ASYNCcnt-tracked) when available.
__device__ __forceinline__ void copy16(const u16* g, u16* l) {
#if HAVE_ASYNC
  __builtin_amdgcn_global_load_async_to_lds_b128(
      (__attribute__((address_space(1))) v4i*)g,
      (__attribute__((address_space(3))) v4i*)l, 0, 0);
#else
  *(int4*)l = *(const int4*)g;
#endif
}

// ---- elementwise converts ----
__global__ void cvt_bf16(const float* __restrict__ in, u16* __restrict__ out, int n) {
  int id = blockIdx.x * 256 + threadIdx.x;
  if (id < n) out[id] = f32_to_bf16(in[id]);
}

// W [K_, N_] f32 row-major  ->  WT [N_, K_] bf16 (transposed)
__global__ void cvt_T(const float* __restrict__ W, u16* __restrict__ WT, int K_, int N_) {
  int id = blockIdx.x * 256 + threadIdx.x;
  if (id < K_ * N_) {
    int k = id / N_;
    int n = id - k * N_;
    WT[(size_t)n * K_ + k] = f32_to_bf16(W[id]);
  }
}

// ---- LDS-staged double-buffered WMMA GEMM ----
// C[M,N] f32 = A[M,K] bf16 @ BT[N,K] bf16
// block = 256 threads (8 waves, 2x4), block tile 64x128, K-step 64.
// Operands staged through LDS with async global->LDS copies (ASYNCcnt),
// 6 async ops per wave per buffer; padded LDS rows (72 halfs) keep 16B
// alignment for ds_load_b128 fragments and spread banks.
__global__ __launch_bounds__(256) void gemm_bf16(const u16* __restrict__ A,
                                                 const u16* __restrict__ BT,
                                                 float* __restrict__ C,
                                                 int Mdim, int Ndim, int Kdim) {
  __shared__ u16 As[2][64][72];
  __shared__ u16 Bs[2][128][72];

  const int tid = threadIdx.x;
  const int lane = tid & 31;
  const int wave = tid >> 5;
  const int lr = lane & 15;
  const int lh = lane >> 4;
  const int wr = (wave >> 2) * 32;     // wave row offset in block tile
  const int wc = (wave & 3) * 32;      // wave col offset in block tile
  const int rowBlock = blockIdx.y * 64;
  const int colBlock = blockIdx.x * 128;

  const u16* Ag = A + (size_t)rowBlock * Kdim;
  const u16* Bg = BT + (size_t)colBlock * Kdim;

  auto stage = [&](int buf, int kk) {
#pragma unroll
    for (int j = 0; j < 2; ++j) {           // A tile: 64x64 = 512 16B slots
      int s = tid + j * 256;
      int row = s >> 3, ch = (s & 7) * 8;
      copy16(Ag + (size_t)row * Kdim + kk + ch, &As[buf][row][ch]);
    }
#pragma unroll
    for (int j = 0; j < 4; ++j) {           // B tile: 128x64 = 1024 16B slots
      int s = tid + j * 256;
      int row = s >> 3, ch = (s & 7) * 8;
      copy16(Bg + (size_t)row * Kdim + kk + ch, &Bs[buf][row][ch]);
    }
  };

  v8f acc[2][2];
#pragma unroll
  for (int i = 0; i < 2; ++i)
#pragma unroll
    for (int j = 0; j < 2; ++j) acc[i][j] = vzero8();

  const int nk = Kdim >> 6;                 // K-steps of 64
  stage(0, 0);

  for (int kt = 0; kt < nk; ++kt) {
    const int cur = kt & 1;
    if (kt + 1 < nk) {
      stage(cur ^ 1, (kt + 1) << 6);
#if HAVE_ASYNC
      __builtin_amdgcn_s_wait_asynccnt(6); // current buffer's 6 ops done
#endif
    } else {
#if HAVE_ASYNC
      __builtin_amdgcn_s_wait_asynccnt(0);
#endif
    }
    __syncthreads();

#pragma unroll
    for (int kk = 0; kk < 64; kk += 32) {
      v16bf a0 = ld_ab_frag(&As[cur][wr + lr][kk], lh);
      v16bf a1 = ld_ab_frag(&As[cur][wr + 16 + lr][kk], lh);
      v16bf b0 = ld_ab_frag(&Bs[cur][wc + lr][kk], lh);
      v16bf b1 = ld_ab_frag(&Bs[cur][wc + 16 + lr][kk], lh);
      acc[0][0] = wmma_bf16(a0, b0, acc[0][0]);
      acc[0][1] = wmma_bf16(a0, b1, acc[0][1]);
      acc[1][0] = wmma_bf16(a1, b0, acc[1][0]);
      acc[1][1] = wmma_bf16(a1, b1, acc[1][1]);
    }
    __syncthreads();
  }

  // C/D layout: VGPR e -> row (lh*8 + e), col lr (per 16x16 tile)
#pragma unroll
  for (int ii = 0; ii < 2; ++ii)
#pragma unroll
    for (int jj = 0; jj < 2; ++jj)
#pragma unroll
      for (int e = 0; e < 8; ++e) {
        int row = rowBlock + wr + 16 * ii + (lh << 3) + e;
        int col = colBlock + wc + 16 * jj + lr;
        C[(size_t)row * Ndim + col] = acc[ii][jj][e];
      }
  (void)Mdim;
}

// ---- RMSNorm + RoPE + layout kernel ----
// grid = (T, 48): slot s: [0,32) Q head, [32,40) K group, [40,48) V group
// outputs: qbf [H,T,HD] (pre-scaled by 1/sqrt(HD)), kbf [G,T,HD], vt [G,HD,T]
__global__ __launch_bounds__(128) void norm_rope(const float* __restrict__ qkv,
                                                 const float* __restrict__ cs,
                                                 const float* __restrict__ sn,
                                                 const float* __restrict__ qnw,
                                                 const float* __restrict__ knw,
                                                 u16* __restrict__ qbf,
                                                 u16* __restrict__ kbf,
                                                 u16* __restrict__ vt) {
  __shared__ float red[128];
  __shared__ float su[128];
  const int t = blockIdx.x, s = blockIdx.y, d = threadIdx.x;
  const float* row = qkv + (size_t)t * NQKV;

  if (s >= 40) {                       // V: transpose-store, no norm/rope
    int g = s - 40;
    float u = row[HH * HDD + GG * HDD + g * HDD + d];
    vt[(size_t)(g * HDD + d) * TT + t] = f32_to_bf16(u);
    return;
  }
  const bool isq = (s < 32);
  const int off = isq ? s * HDD : HH * HDD + (s - 32) * HDD;
  float u = row[off + d];
  red[d] = u * u;
  __syncthreads();
#pragma unroll
  for (int o2 = 64; o2 > 0; o2 >>= 1) {
    if (d < o2) red[d] += red[d + o2];
    __syncthreads();
  }
  float rr = rsqrtf(red[0] * (1.0f / HDD) + 1e-6f);
  float un = u * rr * (isq ? qnw[d] : knw[d]);
  su[d] = un;
  __syncthreads();
  float rot = (d < HDD / 2) ? -su[d + HDD / 2] : su[d - HDD / 2];
  float val = un * cs[(size_t)t * HDD + d] + rot * sn[(size_t)t * HDD + d];
  if (isq) {
    val *= 0.08838834764831845f;       // 1/sqrt(HD) folded into q
    qbf[((size_t)(s * TT + t)) * HDD + d] = f32_to_bf16(val);
  } else {
    kbf[((size_t)((s - 32) * TT + t)) * HDD + d] = f32_to_bf16(val);
  }
}

// ---- causal flash attention ----
// 1 wave = 16 query rows x HD=128; kv tiles of 32; online softmax.
// grid = H*(T/16)/8 blocks of 256 threads (8 independent waves).
__global__ __launch_bounds__(256) void attention(const u16* __restrict__ qbf,
                                                 const u16* __restrict__ kbf,
                                                 const u16* __restrict__ vt,
                                                 u16* __restrict__ attnbf) {
  __shared__ u16 Pld[8][16 * 32];      // per-wave P staging (C->A relayout)
  const int lane = threadIdx.x & 31;
  const int wave = threadIdx.x >> 5;
  const int tile = blockIdx.x * 8 + wave;
  const int h = tile >> 7;             // / (T/16)
  const int q0 = (tile & 127) << 4;
  const int g = h >> 2;                // GQA: GS = H/G = 4
  const int lr = lane & 15;
  const int lh = lane >> 4;
  const int rowBase = q0 + (lh << 3);

  // Q fragments (16 rows x 128 dims = 4 K-chunks), loaded once
  const u16* qrow = qbf + (size_t)(h * TT + q0 + lr) * HDD;
  v16bf qf[4];
#pragma unroll
  for (int f = 0; f < 4; ++f) qf[f] = ld_ab_frag(qrow + f * 32, lh);

  v8f o[8];
#pragma unroll
  for (int j = 0; j < 8; ++j) o[j] = vzero8();
  float m[8], l[8];
#pragma unroll
  for (int i = 0; i < 8; ++i) { m[i] = -1e30f; l[i] = 0.0f; }

  const u16* kb = kbf + (size_t)g * TT * HDD;
  const u16* vb = vt + (size_t)g * HDD * TT;
  u16* P = Pld[wave];

  for (int kv0 = 0; kv0 <= q0 + 15; kv0 += 32) {
    // prefetch next kv tile of K and V streams (global_prefetch_b8)
    if (kv0 + 32 <= q0 + 15) {
      __builtin_prefetch(kb + (size_t)(kv0 + 32 + lr) * HDD, 0, 1);
      __builtin_prefetch(vb + (size_t)lr * TT + kv0 + 32, 0, 1);
    }
    // S = Q @ K^T  (two 16x16 score tiles over 32 kv cols)
    v8f s0 = vzero8(), s1 = vzero8();
#pragma unroll
    for (int f = 0; f < 4; ++f) {
      v16bf b0 = ld_ab_frag(kb + (size_t)(kv0 + lr) * HDD + f * 32, lh);
      v16bf b1 = ld_ab_frag(kb + (size_t)(kv0 + 16 + lr) * HDD + f * 32, lh);
      s0 = wmma_bf16(qf[f], b0, s0);
      s1 = wmma_bf16(qf[f], b1, s1);
    }
    // causal mask in C-fragment coordinates
    const int c0 = kv0 + lr;
#pragma unroll
    for (int i = 0; i < 8; ++i) {
      int row = rowBase + i;
      if (c0 > row)      s0[i] = -1e30f;
      if (c0 + 16 > row) s1[i] = -1e30f;
    }
    // online softmax: row max / exp / row sum (xor-shuffles stay in 16-lane half)
    float alpha[8];
#pragma unroll
    for (int i = 0; i < 8; ++i) {
      float t = fmaxf(s0[i], s1[i]);
      t = fmaxf(t, __shfl_xor(t, 1));
      t = fmaxf(t, __shfl_xor(t, 2));
      t = fmaxf(t, __shfl_xor(t, 4));
      t = fmaxf(t, __shfl_xor(t, 8));
      float mn = fmaxf(m[i], t);
      alpha[i] = __expf(m[i] - mn);
      m[i] = mn;
      float p0 = __expf(s0[i] - mn);
      float p1 = __expf(s1[i] - mn);
      s0[i] = p0; s1[i] = p1;
      float rs = p0 + p1;
      rs += __shfl_xor(rs, 1);
      rs += __shfl_xor(rs, 2);
      rs += __shfl_xor(rs, 4);
      rs += __shfl_xor(rs, 8);
      l[i] = l[i] * alpha[i] + rs;
    }
#pragma unroll
    for (int j = 0; j < 8; ++j)
#pragma unroll
      for (int i = 0; i < 8; ++i) o[j][i] *= alpha[i];

    // P: C-layout -> LDS -> A-layout (wave-local; DS ops are in-order per wave)
#pragma unroll
    for (int i = 0; i < 8; ++i) {
      P[(size_t)((lh << 3) + i) * 32 + lr]      = f32_to_bf16(s0[i]);
      P[(size_t)((lh << 3) + i) * 32 + lr + 16] = f32_to_bf16(s1[i]);
    }
    __builtin_amdgcn_wave_barrier();
    v16bf pa = ld_ab_frag(P + lr * 32, lh);

    // O += P @ V   (V stored transposed: vt[g][d][t])
#pragma unroll
    for (int j = 0; j < 8; ++j) {
      v16bf bv = ld_ab_frag(vb + (size_t)(j * 16 + lr) * TT + kv0, lh);
      o[j] = wmma_bf16(pa, bv, o[j]);
    }
    __builtin_amdgcn_wave_barrier();
  }

  float inv[8];
#pragma unroll
  for (int i = 0; i < 8; ++i) inv[i] = 1.0f / l[i];
#pragma unroll
  for (int j = 0; j < 8; ++j)
#pragma unroll
    for (int i = 0; i < 8; ++i) {
      int row = rowBase + i;
      int col = h * HDD + j * 16 + lr;
      attnbf[(size_t)row * (HH * HDD) + col] = f32_to_bf16(o[j][i] * inv[i]);
    }
}

extern "C" void kernel_launch(void* const* d_in, const int* in_sizes, int n_in,
                              void* d_out, int out_size, void* d_ws, size_t ws_size,
                              hipStream_t stream) {
  (void)in_sizes; (void)n_in; (void)out_size; (void)ws_size;
  const float* x    = (const float*)d_in[0];
  // d_in[1] = mask (unused; causal mask computed analytically)
  const float* cosr = (const float*)d_in[2];
  const float* sinr = (const float*)d_in[3];
  const float* Wq   = (const float*)d_in[4];
  const float* Wk   = (const float*)d_in[5];
  const float* Wv   = (const float*)d_in[6];
  const float* Wo   = (const float*)d_in[7];
  const float* qnw  = (const float*)d_in[8];
  const float* knw  = (const float*)d_in[9];
  float* out = (float*)d_out;

  // workspace layout (~184 MB)
  u16* xbf    = (u16*)d_ws;                          // T*D bf16
  u16* wqkvT  = xbf + (size_t)TT * DD;               // [6144, 4096] bf16 (transposed QKV weights)
  u16* woT    = wqkvT + (size_t)NQKV * DD;           // [4096, 4096] bf16
  float* qkv  = (float*)(woT + (size_t)DD * DD);     // [T, 6144] f32
  u16* qbf    = (u16*)(qkv + (size_t)TT * NQKV);     // [H,T,HD]
  u16* kbf    = qbf + (size_t)HH * TT * HDD;         // [G,T,HD]
  u16* vtbf   = kbf + (size_t)GG * TT * HDD;         // [G,HD,T]
  u16* attnbf = vtbf + (size_t)GG * TT * HDD;        // [T, H*HD]

  cvt_bf16<<<(TT * DD) / 256, 256, 0, stream>>>(x, xbf, TT * DD);
  cvt_T<<<(DD * HH * HDD) / 256, 256, 0, stream>>>(Wq, wqkvT, DD, HH * HDD);
  cvt_T<<<(DD * GG * HDD) / 256, 256, 0, stream>>>(Wk, wqkvT + (size_t)(HH * HDD) * DD, DD, GG * HDD);
  cvt_T<<<(DD * GG * HDD) / 256, 256, 0, stream>>>(Wv, wqkvT + (size_t)(HH * HDD + GG * HDD) * DD, DD, GG * HDD);
  cvt_T<<<(DD * DD) / 256, 256, 0, stream>>>(Wo, woT, HH * HDD, DD);

  { dim3 g(NQKV / 128, TT / 64);
    gemm_bf16<<<g, 256, 0, stream>>>(xbf, wqkvT, qkv, TT, NQKV, DD); }

  { dim3 g(TT, 48);
    norm_rope<<<g, 128, 0, stream>>>(qkv, cosr, sinr, qnw, knw, qbf, kbf, vtbf); }

  attention<<<(HH * (TT / 16)) / 8, 256, 0, stream>>>(qbf, kbf, vtbf, attnbf);

  { dim3 g(DD / 128, TT / 64);
    gemm_bf16<<<g, 256, 0, stream>>>(attnbf, woT, out, TT, DD, HH * HDD); }
}
